// CabinetEncoder_64828236366715
// MI455X (gfx1250) — compile-verified, matrix operand order checked by reading, not run on hardware
//
#include <hip/hip_runtime.h>

typedef __bf16 bf16_t;
typedef bf16_t bf16x8 __attribute__((ext_vector_type(8)));
typedef bf16_t v16bf  __attribute__((ext_vector_type(16)));
typedef float  v8f    __attribute__((ext_vector_type(8)));
typedef float  v4f    __attribute__((ext_vector_type(4)));

#define K_DIM 512
#define N_DIM 512
#define M_DIM 32768

// ---------------------------------------------------------------------------
// Prep: W2 [K][N] f32 (row-major, k-major) -> W2^T [N][K] bf16 split hi/lo.
// Makes WMMA B-fragment loads contiguous 16B per lane.
// ---------------------------------------------------------------------------
__global__ __launch_bounds__(256) void cab_prep_w2(
    const float* __restrict__ W2,
    bf16_t* __restrict__ w2t_hi,
    bf16_t* __restrict__ w2t_lo) {
  int idx = blockIdx.x * 256 + threadIdx.x;   // 0 .. 512*512-1
  int k = idx >> 9;
  int n = idx & (N_DIM - 1);
  float w  = W2[idx];
  bf16_t h = (bf16_t)w;                       // RNE truncation to bf16
  float  r = w - (float)h;                    // residual
  w2t_hi[n * K_DIM + k] = h;
  w2t_lo[n * K_DIM + k] = (bf16_t)r;
}

// ---------------------------------------------------------------------------
// Fused gather + bias + relu + split-bf16 GEMM (M=32768, N=512, K=512)
// Block: 256 thr = 8 waves as 4(M) x 2(N); wave tile 32x64; block tile 128x128.
// Accumulate f32 via: acc += Al*Bh + Ah*Bl + Ah*Bh   (~f32 fidelity)
// ---------------------------------------------------------------------------
__global__ __launch_bounds__(256) void cab_gemm(
    const int*    __restrict__ xidx,
    const float*  __restrict__ W1,
    const float*  __restrict__ b1,
    const bf16_t* __restrict__ w2t_hi,
    const bf16_t* __restrict__ w2t_lo,
    const float*  __restrict__ b2,
    float*        __restrict__ out) {
  const int lane   = threadIdx.x & 31;
  const int wave   = threadIdx.x >> 5;   // 0..7
  const int wm     = wave & 3;           // wave position in M
  const int wn     = wave >> 2;          // wave position in N (0..1)
  const int laneLo = lane & 15;
  const int laneHi = lane >> 4;          // 0 or 1 (half-wave K offset)

  const int M0 = blockIdx.x * 128 + wm * 32;
  const int N0 = blockIdx.y * 128 + wn * 64;

  v8f acc[2][4];
#pragma unroll
  for (int i = 0; i < 2; ++i)
#pragma unroll
    for (int j = 0; j < 4; ++j) { v8f z = {}; acc[i][j] = z; }

  // Per-lane gathered A row pointers (hoisted out of K loop).
  const float* aptr[2];
#pragma unroll
  for (int mt = 0; mt < 2; ++mt)
    aptr[mt] = W1 + (size_t)xidx[M0 + mt * 16 + laneLo] * K_DIM;

  // Per-lane B row pointers (W2^T, row-major N x K).
  const bf16_t* bhp[4];
  const bf16_t* blp[4];
#pragma unroll
  for (int nt = 0; nt < 4; ++nt) {
    size_t o = (size_t)(N0 + nt * 16 + laneLo) * K_DIM;
    bhp[nt] = w2t_hi + o;
    blp[nt] = w2t_lo + o;
  }

  for (int kb = 0; kb < K_DIM; kb += 32) {
    // Per-lane K base: lanes 0-15 take K = kb+0..7 & kb+16..23,
    //                  lanes 16-31 take K = kb+8..15 & kb+24..31 (ISA layout).
    const int r0 = kb + (laneHi << 3);

    // bias chunk (shared by both M tiles)
    v4f bb0 = *(const v4f*)(b1 + r0);
    v4f bb1 = *(const v4f*)(b1 + r0 + 4);
    v4f bb2 = *(const v4f*)(b1 + r0 + 16);
    v4f bb3 = *(const v4f*)(b1 + r0 + 20);

    // ---- A fragments: gather f32, +b1, relu, split to bf16 hi/lo ----
    v16bf aHi[2], aLo[2];
#pragma unroll
    for (int mt = 0; mt < 2; ++mt) {
      const float* ap = aptr[mt];
      v4f a0 = *(const v4f*)(ap + r0);
      v4f a1 = *(const v4f*)(ap + r0 + 4);
      v4f a2 = *(const v4f*)(ap + r0 + 16);
      v4f a3 = *(const v4f*)(ap + r0 + 20);
      if (kb + 32 < K_DIM) __builtin_prefetch(ap + r0 + 32, 0, 0);

      float h[16];
#pragma unroll
      for (int j = 0; j < 4; ++j) {
        h[j]      = fmaxf(a0[j] + bb0[j], 0.0f);
        h[4 + j]  = fmaxf(a1[j] + bb1[j], 0.0f);
        h[8 + j]  = fmaxf(a2[j] + bb2[j], 0.0f);
        h[12 + j] = fmaxf(a3[j] + bb3[j], 0.0f);
      }
      v16bf hi, lo;
#pragma unroll
      for (int j = 0; j < 16; ++j) {
        bf16_t t = (bf16_t)h[j];
        hi[j] = t;
        lo[j] = (bf16_t)(h[j] - (float)t);
      }
      aHi[mt] = hi;
      aLo[mt] = lo;
    }

    // ---- B fragments: contiguous 16B loads of pre-transposed bf16 ----
    v16bf bHi[4], bLo[4];
#pragma unroll
    for (int nt = 0; nt < 4; ++nt) {
      bf16x8 p0 = *(const bf16x8*)(bhp[nt] + r0);
      bf16x8 p1 = *(const bf16x8*)(bhp[nt] + r0 + 16);
      bHi[nt] = __builtin_shufflevector(p0, p1, 0, 1, 2, 3, 4, 5, 6, 7,
                                        8, 9, 10, 11, 12, 13, 14, 15);
      bf16x8 q0 = *(const bf16x8*)(blp[nt] + r0);
      bf16x8 q1 = *(const bf16x8*)(blp[nt] + r0 + 16);
      bLo[nt] = __builtin_shufflevector(q0, q1, 0, 1, 2, 3, 4, 5, 6, 7,
                                        8, 9, 10, 11, 12, 13, 14, 15);
    }

    // ---- 24 WMMAs per K-step: low-order correction terms first ----
#pragma unroll
    for (int mt = 0; mt < 2; ++mt)
#pragma unroll
      for (int nt = 0; nt < 4; ++nt) {
        acc[mt][nt] = __builtin_amdgcn_wmma_f32_16x16x32_bf16(
            false, aLo[mt], false, bHi[nt], (short)0, acc[mt][nt], false, false);
        acc[mt][nt] = __builtin_amdgcn_wmma_f32_16x16x32_bf16(
            false, aHi[mt], false, bLo[nt], (short)0, acc[mt][nt], false, false);
        acc[mt][nt] = __builtin_amdgcn_wmma_f32_16x16x32_bf16(
            false, aHi[mt], false, bHi[nt], (short)0, acc[mt][nt], false, false);
      }
  }

  // ---- Epilogue: + b2, f32 store.
  // C/D layout: lane 0-15 -> col=laneLo, rows v+0..7; lane 16-31 -> rows v+8..15.
#pragma unroll
  for (int mt = 0; mt < 2; ++mt) {
    const int rowBase = M0 + mt * 16 + (laneHi << 3);
#pragma unroll
    for (int nt = 0; nt < 4; ++nt) {
      const int col = N0 + nt * 16 + laneLo;
      const float bias2 = b2[col];
#pragma unroll
      for (int v = 0; v < 8; ++v)
        out[(size_t)(rowBase + v) * N_DIM + col] = acc[mt][nt][v] + bias2;
    }
  }
}

extern "C" void kernel_launch(void* const* d_in, const int* in_sizes, int n_in,
                              void* d_out, int out_size, void* d_ws, size_t ws_size,
                              hipStream_t stream) {
  (void)in_sizes; (void)n_in; (void)out_size; (void)ws_size;
  const int*   x  = (const int*)d_in[0];
  const float* W1 = (const float*)d_in[1];
  const float* b1 = (const float*)d_in[2];
  const float* W2 = (const float*)d_in[3];
  const float* b2 = (const float*)d_in[4];
  float* out = (float*)d_out;

  bf16_t* w2t_hi = (bf16_t*)d_ws;                  // 512*512 bf16 = 512 KB
  bf16_t* w2t_lo = w2t_hi + (size_t)K_DIM * N_DIM; // +512 KB

  cab_prep_w2<<<(K_DIM * N_DIM) / 256, 256, 0, stream>>>(W2, w2t_hi, w2t_lo);

  dim3 grid(M_DIM / 128, N_DIM / 128);
  cab_gemm<<<grid, 256, 0, stream>>>(x, W1, b1, w2t_hi, w2t_lo, b2, out);
}
